// cross_attn_68788196213006
// MI455X (gfx1250) — compile-verified, matrix operand order checked by reading, not run on hardware
//
#include <hip/hip_runtime.h>
#include <hip/hip_bf16.h>

#define DIM      1024
#define HEADS    16
#define HEAD_DIM 64
#define NTOK     4096
#define EPS      1e-5f
#define SCALE    0.03125f   // DIM^-0.5

typedef __attribute__((ext_vector_type(16))) _Float16 v16h;
typedef __attribute__((ext_vector_type(8)))  _Float16 v8h;
typedef __attribute__((ext_vector_type(4)))  _Float16 v4h;
typedef __attribute__((ext_vector_type(8)))  float    v8f;
typedef __attribute__((ext_vector_type(4)))  float    v4f;

// Pack two 8-half chunks (16B each) into a WMMA 16x32 operand register set.
static __device__ __forceinline__ v16h pack16(const _Float16* plo, const _Float16* phi) {
  v8h lo = *(const v8h*)plo;
  v8h hi = *(const v8h*)phi;
  v16h o;
#pragma unroll
  for (int j = 0; j < 8; ++j) { o[j] = lo[j]; o[8 + j] = hi[j]; }
  return o;
}

// Build a WMMA 16x32 f16 operand from a row-major f16 matrix in global memory.
// Lane l: row = l&15, K in [kh, kh+8) and [16+kh, 16+kh+8), kh = (l>>4)*8.
// Used for A (rows = M) and for B fed as B^T rows (rows = N, K-contiguous).
static __device__ __forceinline__ v16h wmma_operand_global(const _Float16* base, int ld,
                                                           int row0, int k0) {
  const int lane = threadIdx.x & 31;
  const int r    = lane & 15;
  const int kh   = (lane >> 4) << 3;
  const _Float16* p = base + (size_t)(row0 + r) * ld + k0 + kh;
  return pack16(p, p + 16);
}

// ---------------- LayerNorm: f32 row -> f16 row ----------------
__global__ void ln_kernel(const float* __restrict__ x, const float* __restrict__ g,
                          const float* __restrict__ b, _Float16* __restrict__ out) {
  __shared__ float red[256];
  const int row = blockIdx.x;
  const int t = threadIdx.x;
  const float* xr = x + (size_t)row * DIM;
  v4f v = *(const v4f*)(xr + t * 4);
  red[t] = v[0] + v[1] + v[2] + v[3];
  __syncthreads();
  for (int o = 128; o > 0; o >>= 1) { if (t < o) red[t] += red[t + o]; __syncthreads(); }
  const float mu = red[0] * (1.0f / DIM);
  __syncthreads();
  float d0 = v[0] - mu, d1 = v[1] - mu, d2 = v[2] - mu, d3 = v[3] - mu;
  red[t] = d0 * d0 + d1 * d1 + d2 * d2 + d3 * d3;
  __syncthreads();
  for (int o = 128; o > 0; o >>= 1) { if (t < o) red[t] += red[t + o]; __syncthreads(); }
  const float rs = rsqrtf(red[0] * (1.0f / DIM) + EPS);
  _Float16* orow = out + (size_t)row * DIM;
#pragma unroll
  for (int j = 0; j < 4; ++j) {
    const int c = t * 4 + j;
    orow[c] = (_Float16)((v[j] - mu) * rs * g[c] + b[c]);
  }
}

// ---------------- f32 -> f16 weight conversion ----------------
__global__ void cvt_kernel(const float* __restrict__ src, _Float16* __restrict__ dst) {
  const int i = (blockIdx.x * 256 + threadIdx.x) * 4;
  v4f v = *(const v4f*)(src + i);
#pragma unroll
  for (int j = 0; j < 4; ++j) dst[i + j] = (_Float16)v[j];
}

// ---------------- Projection GEMM: out = X @ W^T + bias (f16 out) ----------------
// block = 128 threads (4 waves); wave w does 16x16 tile at (blockIdx.x*16, blockIdx.y*64 + w*16)
__global__ void proj_gemm(const _Float16* __restrict__ X, const _Float16* __restrict__ W,
                          const float* __restrict__ bias, _Float16* __restrict__ out) {
  const int wave = threadIdx.x >> 5;
  const int m0 = blockIdx.x * 16;
  const int c0 = blockIdx.y * 64 + wave * 16;
  v8f acc = {};
  for (int k = 0; k < DIM; k += 32) {
    v16h a = wmma_operand_global(X, DIM, m0, k);  // rows of X
    v16h b = wmma_operand_global(W, DIM, c0, k);  // rows of W == rows of (W^T)^T
    acc = __builtin_amdgcn_wmma_f32_16x16x32_f16(false, a, false, b, (short)0, acc, false, false);
  }
  const int lane = threadIdx.x & 31;
  const int nn = lane & 15;
  const int mb = (lane >> 4) << 3;
  const float bv = bias[c0 + nn];
#pragma unroll
  for (int r = 0; r < 8; ++r)
    out[(size_t)(m0 + mb + r) * DIM + c0 + nn] = (_Float16)(acc[r] + bv);
}

// ---------------- Scores: sim = scale * q @ k^T per head -> f32 attn buffer ----------------
// block = 256 threads (8 waves); block owns 16 rows of one head; waves sweep 4096 cols.
__global__ void score_kernel(const _Float16* __restrict__ q, const _Float16* __restrict__ k,
                             float* __restrict__ attn) {
  const int wave = threadIdx.x >> 5;
  const int h = blockIdx.y;
  const int m0 = blockIdx.x * 16;
  const int koff = h * HEAD_DIM;
  const v16h a0 = wmma_operand_global(q, DIM, m0, koff);
  const v16h a1 = wmma_operand_global(q, DIM, m0, koff + 32);
  float* aH = attn + (size_t)h * NTOK * NTOK;
  const int lane = threadIdx.x & 31;
  const int nn = lane & 15;
  const int mb = (lane >> 4) << 3;
  for (int c0 = wave * 16; c0 < NTOK; c0 += 8 * 16) {
    v16h b0 = wmma_operand_global(k, DIM, c0, koff);       // B^T rows = k rows
    v16h b1 = wmma_operand_global(k, DIM, c0, koff + 32);
    v8f acc = {};
    acc = __builtin_amdgcn_wmma_f32_16x16x32_f16(false, a0, false, b0, (short)0, acc, false, false);
    acc = __builtin_amdgcn_wmma_f32_16x16x32_f16(false, a1, false, b1, (short)0, acc, false, false);
#pragma unroll
    for (int r = 0; r < 8; ++r)
      aH[(size_t)(m0 + mb + r) * NTOK + c0 + nn] = acc[r] * SCALE;
  }
}

// ---------------- Row softmax over 4096 elements, in place ----------------
__global__ void softmax_kernel(float* __restrict__ attn) {
  __shared__ float red[256];
  const int t = threadIdx.x;
  float* row = attn + (size_t)blockIdx.x * NTOK;
  float vals[16];
  float m = -1e30f;
#pragma unroll
  for (int i = 0; i < 4; ++i) {
    v4f v = *(const v4f*)(row + i * 1024 + t * 4);
#pragma unroll
    for (int j = 0; j < 4; ++j) { vals[i * 4 + j] = v[j]; m = fmaxf(m, v[j]); }
  }
  red[t] = m; __syncthreads();
  for (int o = 128; o > 0; o >>= 1) { if (t < o) red[t] = fmaxf(red[t], red[t + o]); __syncthreads(); }
  const float rmax = red[0]; __syncthreads();
  float s = 0.0f;
#pragma unroll
  for (int e = 0; e < 16; ++e) { vals[e] = __expf(vals[e] - rmax); s += vals[e]; }
  red[t] = s; __syncthreads();
  for (int o = 128; o > 0; o >>= 1) { if (t < o) red[t] += red[t + o]; __syncthreads(); }
  const float inv = 1.0f / red[0];
#pragma unroll
  for (int i = 0; i < 4; ++i) {
    v4f o;
#pragma unroll
    for (int j = 0; j < 4; ++j) o[j] = vals[i * 4 + j] * inv;
    *(v4f*)(row + i * 1024 + t * 4) = o;
  }
}

// ---------------- attn GEMM: out = (TRANSA ? attn^T : attn) @ X  per head ----------------
// block = 256 threads (8 waves). Block computes 32 out-rows x 64 cols (whole head).
// Wave w: row-group g = w>>2 (rows g*16..g*16+15), col-group cw = w&3 (cols cw*16..+15).
// K loop over 4096 in steps of 64 (2 WMMAs per wave per barrier pair).
// attn tile (f32->f16) -> As[m][k] (ld 72); X tile -> XsT[c][k] (ld 72, K-contiguous)
// so every WMMA operand is two 16B ds_load_b128 per lane.
#define LDK 72
template <int TRANSA>
__global__ void attn_gemm(const float* __restrict__ attn, const _Float16* __restrict__ X,
                          float* __restrict__ out) {
  __shared__ _Float16 As[32 * LDK];    // As[m][k], m<32, k<64
  __shared__ _Float16 XsT[64 * LDK];   // XsT[c][k], c<64, k<64
  const int t = threadIdx.x;
  const int wave = t >> 5;
  const int lane = t & 31;
  const int g16 = (wave >> 2) << 4;    // row group offset
  const int cw16 = (wave & 3) << 4;    // col group offset
  const int h = blockIdx.y;
  const int m0 = blockIdx.x * 32;
  const float* aH = attn + (size_t)h * NTOK * NTOK;
  const _Float16* xH = X + h * HEAD_DIM;
  const int r  = lane & 15;
  const int kh = (lane >> 4) << 3;
  v8f acc = {};
  for (int k0 = 0; k0 < NTOK; k0 += 64) {
    // stage X rows k0..k0+63 transposed: XsT[c][k] = X[k0+k][h*64+c]
#pragma unroll
    for (int i = 0; i < 2; ++i) {
      const int cid = t + i * 256;              // 512 chunks of 8 halves
      const int kr = cid >> 3, c8 = (cid & 7) << 3;
      v8h v = *(const v8h*)(xH + (size_t)(k0 + kr) * DIM + c8);
#pragma unroll
      for (int j = 0; j < 8; ++j) XsT[(c8 + j) * LDK + kr] = v[j];
    }
    // stage attn tile as As[m][k] (f32 -> f16)
    if (TRANSA == 0) {
#pragma unroll
      for (int i = 0; i < 2; ++i) {
        const int aid = t + i * 256;            // 512 chunks of 4 floats
        const int mr = aid >> 4, kk = (aid & 15) << 2;
        v4f v = *(const v4f*)(aH + (size_t)(m0 + mr) * NTOK + k0 + kk);
        v4h hv;
#pragma unroll
        for (int j = 0; j < 4; ++j) hv[j] = (_Float16)v[j];
        *(v4h*)&As[mr * LDK + kk] = hv;
      }
    } else {
#pragma unroll
      for (int i = 0; i < 2; ++i) {
        const int aid = t + i * 256;
        const int kr = aid >> 3, mm = (aid & 7) << 2;
        v4f v = *(const v4f*)(aH + (size_t)(k0 + kr) * NTOK + m0 + mm);
#pragma unroll
        for (int j = 0; j < 4; ++j) As[(mm + j) * LDK + kr] = (_Float16)v[j];
      }
    }
    __syncthreads();
    const _Float16* arow = &As[(g16 + r) * LDK];
    const _Float16* brow = &XsT[(cw16 + r) * LDK];
    v16h a0 = pack16(arow + kh,      arow + kh + 16);
    v16h b0 = pack16(brow + kh,      brow + kh + 16);
    v16h a1 = pack16(arow + kh + 32, arow + kh + 48);
    v16h b1 = pack16(brow + kh + 32, brow + kh + 48);
    acc = __builtin_amdgcn_wmma_f32_16x16x32_f16(false, a0, false, b0, (short)0, acc, false, false);
    acc = __builtin_amdgcn_wmma_f32_16x16x32_f16(false, a1, false, b1, (short)0, acc, false, false);
    __syncthreads();
  }
  const int nn = lane & 15;
  const int mb = (lane >> 4) << 3;
#pragma unroll
  for (int rr = 0; rr < 8; ++rr)
    out[(size_t)(m0 + g16 + mb + rr) * DIM + h * HEAD_DIM + cw16 + nn] = acc[rr];
}

extern "C" void kernel_launch(void* const* d_in, const int* in_sizes, int n_in,
                              void* d_out, int out_size, void* d_ws, size_t ws_size,
                              hipStream_t stream) {
  (void)in_sizes; (void)n_in; (void)out_size; (void)ws_size;
  const float* z1  = (const float*)d_in[0];
  const float* z2  = (const float*)d_in[1];
  const float* wq  = (const float*)d_in[2];
  const float* bq  = (const float*)d_in[3];
  const float* wk  = (const float*)d_in[4];
  const float* bk  = (const float*)d_in[5];
  const float* wv1 = (const float*)d_in[6];
  const float* bv1 = (const float*)d_in[7];
  const float* wv2 = (const float*)d_in[8];
  const float* bv2 = (const float*)d_in[9];
  const float* g1  = (const float*)d_in[10];
  const float* be1 = (const float*)d_in[11];
  const float* g2  = (const float*)d_in[12];
  const float* be2 = (const float*)d_in[13];

  float* z_c  = (float*)d_out;
  float* c_z  = z_c + (size_t)NTOK * DIM;
  float* attn = c_z + (size_t)NTOK * DIM;

  const size_t NT = (size_t)NTOK * DIM;   // 4M halves
  const size_t WT = (size_t)DIM * DIM;    // 1M halves
  _Float16* n1   = (_Float16*)d_ws;
  _Float16* n2   = n1 + NT;
  _Float16* qb   = n2 + NT;
  _Float16* kb   = qb + NT;
  _Float16* vb   = kb + NT;
  _Float16* v1b  = vb + NT;
  _Float16* wqh  = v1b + NT;
  _Float16* wkh  = wqh + WT;
  _Float16* wv1h = wkh + WT;
  _Float16* wv2h = wv1h + WT;

  ln_kernel<<<NTOK, 256, 0, stream>>>(z1, g1, be1, n1);
  ln_kernel<<<NTOK, 256, 0, stream>>>(z2, g2, be2, n2);

  const int cvt_blocks = (int)(WT / 1024);
  cvt_kernel<<<cvt_blocks, 256, 0, stream>>>(wq,  wqh);
  cvt_kernel<<<cvt_blocks, 256, 0, stream>>>(wk,  wkh);
  cvt_kernel<<<cvt_blocks, 256, 0, stream>>>(wv1, wv1h);
  cvt_kernel<<<cvt_blocks, 256, 0, stream>>>(wv2, wv2h);

  dim3 pg(NTOK / 16, DIM / 64);
  proj_gemm<<<pg, 128, 0, stream>>>(n1, wqh,  bq,  qb);
  proj_gemm<<<pg, 128, 0, stream>>>(n2, wkh,  bk,  kb);
  proj_gemm<<<pg, 128, 0, stream>>>(n2, wv1h, bv1, vb);
  proj_gemm<<<pg, 128, 0, stream>>>(n1, wv2h, bv2, v1b);

  dim3 sg(NTOK / 16, HEADS);
  score_kernel<<<sg, 256, 0, stream>>>(qb, kb, attn);

  softmax_kernel<<<HEADS * NTOK, 256, 0, stream>>>(attn);

  dim3 ag(NTOK / 32, HEADS);
  attn_gemm<0><<<ag, 256, 0, stream>>>(attn, vb,  z_c);
  attn_gemm<1><<<ag, 256, 0, stream>>>(attn, v1b, c_z);
}
